// ScaledDotAttention_16398185136587
// MI455X (gfx1250) — compile-verified, matrix operand order checked by reading, not run on hardware
//
#include <hip/hip_runtime.h>
#include <math.h>

// ---------------------------------------------------------------------------
// ScaledDotAttention for MI455X (gfx1250, wave32, WMMA + TDM)
// B=8, T=2048, C=H=O=512.  bf16 WMMA (16x16x32) with f32 accumulation.
// Pipeline:
//   1) LN(q), LN(k) -> bf16 ; cast v -> bf16 ; cast Wq/Wk/Wv/Wo -> bf16
//   2) Q = qn@Wq^T+bq, K = kn@Wk^T+bk (bf16), V = vb@Wv^T+bv stored TRANSPOSED
//   3) scores = Q@K^T * 1/sqrt(H)  -> written straight into d_out weights region
//   4) softmax in-place on weights (f32 out) + bf16 copy for WMMA
//   5) R = P@V  (bf16) ; out = R@Wo^T + bo (f32)
// GEMM: WG tile 64x128, double-buffered LDS staging via Tensor Data Mover,
//       fragments read from LDS with ds_load_b128, 8 WMMA per wave per K-slab.
// ---------------------------------------------------------------------------

typedef __attribute__((ext_vector_type(16))) __bf16         v16bf;
typedef __attribute__((ext_vector_type(8)))  float          v8f;
typedef __attribute__((ext_vector_type(4)))  float          v4f;
typedef __attribute__((ext_vector_type(4)))  unsigned int   v4u;
typedef __attribute__((ext_vector_type(4)))  unsigned short v4h;
typedef __attribute__((ext_vector_type(4)))  unsigned int   u32x4;
typedef __attribute__((ext_vector_type(8)))  int            i32x8;
typedef __attribute__((ext_vector_type(4)))  int            i32x4;

#define BB 8
#define TT 2048
#define CC 512
#define HH 512
#define OO 512

#define KSLAB 64                      // K depth staged per TDM slab
#define A_ROWS 64                     // WG tile M
#define B_ROWS 128                    // WG tile N
#define BUFE (A_ROWS * KSLAB + B_ROWS * KSLAB)   // 12288 ushort per buffer

#if __has_builtin(__builtin_amdgcn_tensor_load_to_lds)
#define USE_TDM 1
#else
#define USE_TDM 0
#endif

__device__ __forceinline__ unsigned short f32_bf16(float f) {
  unsigned int u = __float_as_uint(f);
  unsigned int r = (u + 0x7FFFu + ((u >> 16) & 1u)) >> 16;   // RNE
  return (unsigned short)r;
}

union Frag {                 // one wave-lane's slice of a 16x32 bf16 matrix tile
  v16bf v;
  v4u   u[2];
};

#if USE_TDM
// Issue one 2D TDM tile load: tile_h rows x tile_w bf16 elements, global row
// stride = stride_e elements, packed contiguously into LDS at lds_off (bytes).
// D# layout per CDNA5 ISA 8.3/8.4: count=1, type=2 (image), data_size=1 (2B),
// tensor window == tile (all in-bounds), no padding / iterate / multicast.
__device__ __forceinline__ void tdm_load_2d(unsigned lds_off,
                                            const unsigned short* gsrc,
                                            unsigned tile_w, unsigned tile_h,
                                            unsigned long long stride_e)
{
  unsigned long long ga = (unsigned long long)(size_t)gsrc;   // byte address
  u32x4 g0;
  g0[0] = 1u;                                    // count=1, user-mode D#
  g0[1] = lds_off;                               // lds_addr (bytes)
  g0[2] = (unsigned)(ga & 0xffffffffu);          // global_addr[31:0]
  g0[3] = (unsigned)((ga >> 32) & 0x01ffffffu)   // global_addr[56:32]
        | (2u << 30);                            // type = 2
  i32x8 g1;
  g1[0] = (int)(1u << 16);                       // data_size = 1 (2 bytes)
  g1[1] = (int)((tile_w & 0xffffu) << 16);       // tensor_dim0[15:0]
  g1[2] = (int)((tile_w >> 16) | ((tile_h & 0xffffu) << 16));  // tdim0 hi | tdim1 lo
  g1[3] = (int)((tile_h >> 16) | (tile_w << 16));              // tdim1 hi | tile_dim0
  g1[4] = (int)(tile_h & 0xffffu);               // tile_dim1 ; tile_dim2 = 0
  g1[5] = (int)(stride_e & 0xffffffffu);         // tensor_dim0_stride[31:0]
  g1[6] = (int)((stride_e >> 32) & 0xffffu);     // stride[47:32] ; dim1_stride lo = 0
  g1[7] = 0;
  i32x4 gz = {0, 0, 0, 0};
#if defined(__clang_major__) && __clang_major__ >= 23
  i32x8 gz8 = {0, 0, 0, 0, 0, 0, 0, 0};
  __builtin_amdgcn_tensor_load_to_lds(g0, g1, gz, gz, gz8, 0);
#else
  __builtin_amdgcn_tensor_load_to_lds(g0, g1, gz, gz, 0);
#endif
}
#endif

// ---------------------------------------------------------------------------
// Generic bf16 GEMM:  C[m,n] = scale * sum_k A[m,k] * Bm[n,k]  (+ bias[n])
// A:  M x K bf16 row-major (lda, batched by sA); Bm: N x K bf16 row-major.
// Output f32 (Cf) and/or bf16 (Cb, optionally transposed n-major).
// 256 threads = 8 waves (2 M x 4 N), wave tile 32x32, WG tile 64x128.
// grid = (N/128, M/64, batch)
// ---------------------------------------------------------------------------
__global__ __launch_bounds__(256) void gemm_bf16(
    const unsigned short* __restrict__ A,  int lda, long long sA,
    const unsigned short* __restrict__ Bm, int ldb, long long sB,
    const float* __restrict__ bias, float scale,
    float* __restrict__ Cf, unsigned short* __restrict__ Cb, int transC,
    int ldc, long long sC, int K)
{
  __shared__ unsigned short smem[2 * BUFE];      // 48 KB double buffer

  const int tid  = threadIdx.x;
  const int lane = tid & 31;
  const int wave = tid >> 5;
  const int wm   = wave >> 2;        // 0..1
  const int wn   = wave & 3;         // 0..3
  const int half = lane >> 4;        // K sub-range select (ISA 16-bit layout)
  const int l16  = lane & 15;
  const long long zb = blockIdx.z;

  const unsigned short* Ag = A  + zb * sA + (long long)(blockIdx.y * 64)  * lda;
  const unsigned short* Bg = Bm + zb * sB + (long long)(blockIdx.x * 128) * ldb;

  const int m0 = blockIdx.y * 64  + wm * 32;
  const int n0 = blockIdx.x * 128 + wn * 32;

#if USE_TDM
  const unsigned lds_base = (unsigned)(size_t)&smem[0];  // LDS byte offset
#endif

  // ---- stage(slab) : copy A(64 x KSLAB) and B(128 x KSLAB) into buffer c ----
  auto stage = [&](int c, int k0) {
#if USE_TDM
    if (tid < 32) {                          // one wave issues; EXEC-independent DMA
      tdm_load_2d(lds_base + (unsigned)(c * BUFE) * 2u,
                  Ag + k0, KSLAB, A_ROWS, (unsigned long long)lda);
      tdm_load_2d(lds_base + (unsigned)(c * BUFE + A_ROWS * KSLAB) * 2u,
                  Bg + k0, KSLAB, B_ROWS, (unsigned long long)ldb);
    }
#else
    {   // cooperative fallback: 256 threads, b128 global->LDS
      unsigned short* sa = &smem[c * BUFE];
      unsigned short* sb = &smem[c * BUFE + A_ROWS * KSLAB];
      const int ra = tid >> 2, ka = (tid & 3) * 16;      // A: 16 elems/thread
      *(v4u*)(sa + ra * KSLAB + ka)     = *(const v4u*)(Ag + (long long)ra * lda + k0 + ka);
      *(v4u*)(sa + ra * KSLAB + ka + 8) = *(const v4u*)(Ag + (long long)ra * lda + k0 + ka + 8);
      const int rb = tid >> 1, kb = (tid & 1) * 32;      // B: 32 elems/thread
      #pragma unroll
      for (int j = 0; j < 4; ++j)
        *(v4u*)(sb + rb * KSLAB + kb + 8 * j) =
            *(const v4u*)(Bg + (long long)rb * ldb + k0 + kb + 8 * j);
    }
#endif
  };

  const v8f z = {0.f,0.f,0.f,0.f,0.f,0.f,0.f,0.f};
  v8f acc[2][2] = {{z, z}, {z, z}};

  stage(0, 0);                                  // prologue: fill buffer 0
  int cur = 0;
  for (int k0 = 0; k0 < K; k0 += KSLAB) {
    const bool more = (k0 + KSLAB) < K;
    if (more) stage(cur ^ 1, k0 + KSLAB);       // async-fill next buffer
#if USE_TDM
    if (tid < 32) {
      if (more) __builtin_amdgcn_s_wait_tensorcnt(2);  // current slab (2 ops) done
      else      __builtin_amdgcn_s_wait_tensorcnt(0);
    }
#endif
    __syncthreads();                            // publish current buffer

    const unsigned short* sA = &smem[cur * BUFE];
    const unsigned short* sB = &smem[cur * BUFE + A_ROWS * KSLAB];
    const int ra0 = (wm * 32      + l16) * KSLAB + half * 8;
    const int ra1 = (wm * 32 + 16 + l16) * KSLAB + half * 8;
    const int rb0 = (wn * 32      + l16) * KSLAB + half * 8;
    const int rb1 = (wn * 32 + 16 + l16) * KSLAB + half * 8;

    #pragma unroll
    for (int ks = 0; ks < KSLAB; ks += 32) {
      Frag a0, a1, b0, b1;
      a0.u[0] = *(const v4u*)(sA + ra0 + ks);
      a0.u[1] = *(const v4u*)(sA + ra0 + ks + 16);
      a1.u[0] = *(const v4u*)(sA + ra1 + ks);
      a1.u[1] = *(const v4u*)(sA + ra1 + ks + 16);
      b0.u[0] = *(const v4u*)(sB + rb0 + ks);
      b0.u[1] = *(const v4u*)(sB + rb0 + ks + 16);
      b1.u[0] = *(const v4u*)(sB + rb1 + ks);
      b1.u[1] = *(const v4u*)(sB + rb1 + ks + 16);

      acc[0][0] = __builtin_amdgcn_wmma_f32_16x16x32_bf16(false, a0.v, false, b0.v,
                                                          (short)0, acc[0][0], false, false);
      acc[0][1] = __builtin_amdgcn_wmma_f32_16x16x32_bf16(false, a0.v, false, b1.v,
                                                          (short)0, acc[0][1], false, false);
      acc[1][0] = __builtin_amdgcn_wmma_f32_16x16x32_bf16(false, a1.v, false, b0.v,
                                                          (short)0, acc[1][0], false, false);
      acc[1][1] = __builtin_amdgcn_wmma_f32_16x16x32_bf16(false, a1.v, false, b1.v,
                                                          (short)0, acc[1][1], false, false);
    }
    __syncthreads();                            // all reads done before overwrite
    cur ^= 1;
  }

  // Epilogue. C/D layout: lane -> n = l16 ; vgpr r -> m = r + 8*half (per subtile)
  #pragma unroll
  for (int im = 0; im < 2; ++im) {
    #pragma unroll
    for (int jn = 0; jn < 2; ++jn) {
      const int n  = n0 + jn * 16 + l16;
      const int mb = m0 + im * 16 + half * 8;
      const float bn = bias ? bias[n] : 0.0f;
      float vals[8];
      #pragma unroll
      for (int r = 0; r < 8; ++r) vals[r] = acc[im][jn][r] * scale + bn;

      if (Cf) {
        float* cp = Cf + zb * sC;
        #pragma unroll
        for (int r = 0; r < 8; ++r) cp[(long long)(mb + r) * ldc + n] = vals[r];
      }
      if (Cb) {
        unsigned short* cp = Cb + zb * sC;
        if (transC) {   // Ct[n][m]: r contiguous in m -> single b128 store
          unsigned int p0 = (unsigned int)f32_bf16(vals[0]) | ((unsigned int)f32_bf16(vals[1]) << 16);
          unsigned int p1 = (unsigned int)f32_bf16(vals[2]) | ((unsigned int)f32_bf16(vals[3]) << 16);
          unsigned int p2 = (unsigned int)f32_bf16(vals[4]) | ((unsigned int)f32_bf16(vals[5]) << 16);
          unsigned int p3 = (unsigned int)f32_bf16(vals[6]) | ((unsigned int)f32_bf16(vals[7]) << 16);
          v4u pk = {p0, p1, p2, p3};
          *(v4u*)(cp + (long long)n * ldc + mb) = pk;
        } else {
          #pragma unroll
          for (int r = 0; r < 8; ++r)
            cp[(long long)(mb + r) * ldc + n] = f32_bf16(vals[r]);
        }
      }
    }
  }
}

// ---------------------------------------------------------------------------
// LayerNorm (doLN=1) or plain cast (doLN=0), f32 row of CC -> bf16 row.
// One wave32 per row, 8 rows per 256-thread block.
// ---------------------------------------------------------------------------
__global__ __launch_bounds__(256) void ln_cast(
    const float* __restrict__ x, const float* __restrict__ g,
    const float* __restrict__ bt, unsigned short* __restrict__ y, int doLN)
{
  const int lane = threadIdx.x & 31;
  const long long row = (long long)blockIdx.x * 8 + (threadIdx.x >> 5);
  const float* xr = x + row * CC;

  v4f d[4];
  float s = 0.f, ss = 0.f;
  #pragma unroll
  for (int i = 0; i < 4; ++i) {
    d[i] = *(const v4f*)(xr + 4 * (lane + 32 * i));
    #pragma unroll
    for (int j = 0; j < 4; ++j) { s += d[i][j]; ss += d[i][j] * d[i][j]; }
  }
  #pragma unroll
  for (int off = 16; off >= 1; off >>= 1) {
    s  += __shfl_xor(s,  off, 32);
    ss += __shfl_xor(ss, off, 32);
  }
  const float mean = s * (1.0f / CC);
  const float inv  = rsqrtf(ss * (1.0f / CC) - mean * mean + 1e-5f);

  unsigned short* yr = y + row * CC;
  #pragma unroll
  for (int i = 0; i < 4; ++i) {
    const int e0 = 4 * (lane + 32 * i);
    v4h pk;
    #pragma unroll
    for (int j = 0; j < 4; ++j) {
      float t = d[i][j];
      if (doLN) t = (t - mean) * inv * g[e0 + j] + bt[e0 + j];
      pk[j] = f32_bf16(t);
    }
    *(v4h*)(yr + e0) = pk;
  }
}

// f32 -> bf16 elementwise (weights), 4 elems/thread
__global__ __launch_bounds__(256) void cast_bf16(
    const float* __restrict__ x, unsigned short* __restrict__ y, int n)
{
  const int i = (blockIdx.x * 256 + threadIdx.x) * 4;
  if (i < n) {
    v4f d = *(const v4f*)(x + i);
    v4h pk = {f32_bf16(d[0]), f32_bf16(d[1]), f32_bf16(d[2]), f32_bf16(d[3])};
    *(v4h*)(y + i) = pk;
  }
}

// ---------------------------------------------------------------------------
// Row softmax over TT=2048. One 256-thread block per row, 8 f32/thread.
// Writes normalized f32 in place + bf16 copy for the P@V WMMA.
// ---------------------------------------------------------------------------
__global__ __launch_bounds__(256) void softmax_row(
    float* __restrict__ w, unsigned short* __restrict__ pb)
{
  const long long row = blockIdx.x;
  float*          wr  = w  + row * (long long)TT;
  unsigned short* pr  = pb + row * (long long)TT;
  const int tid  = threadIdx.x;
  const int lane = tid & 31;
  const int wid  = tid >> 5;

  v4f d0 = *(const v4f*)(wr + 4 * tid);
  v4f d1 = *(const v4f*)(wr + 4 * (tid + 256));

  float mx = -3.402823466e38f;
  #pragma unroll
  for (int j = 0; j < 4; ++j) { mx = fmaxf(mx, d0[j]); mx = fmaxf(mx, d1[j]); }
  #pragma unroll
  for (int off = 16; off >= 1; off >>= 1) mx = fmaxf(mx, __shfl_xor(mx, off, 32));

  __shared__ float red[8];
  if (lane == 0) red[wid] = mx;
  __syncthreads();
  if (tid == 0) {
    float m = red[0];
    #pragma unroll
    for (int i = 1; i < 8; ++i) m = fmaxf(m, red[i]);
    red[0] = m;
  }
  __syncthreads();
  mx = red[0];
  __syncthreads();

  float sum = 0.f;
  #pragma unroll
  for (int j = 0; j < 4; ++j) {
    d0[j] = __expf(d0[j] - mx); sum += d0[j];
    d1[j] = __expf(d1[j] - mx); sum += d1[j];
  }
  #pragma unroll
  for (int off = 16; off >= 1; off >>= 1) sum += __shfl_xor(sum, off, 32);
  if (lane == 0) red[wid] = sum;
  __syncthreads();
  if (tid == 0) {
    float t = 0.f;
    #pragma unroll
    for (int i = 1; i < 8; ++i) t += red[i];
    red[0] += t;
  }
  __syncthreads();
  const float inv = 1.0f / red[0];

  v4h p0, p1;
  #pragma unroll
  for (int j = 0; j < 4; ++j) {
    d0[j] *= inv; d1[j] *= inv;
    p0[j] = f32_bf16(d0[j]);
    p1[j] = f32_bf16(d1[j]);
  }
  *(v4f*)(wr + 4 * tid)         = d0;
  *(v4f*)(wr + 4 * (tid + 256)) = d1;
  *(v4h*)(pr + 4 * tid)         = p0;
  *(v4h*)(pr + 4 * (tid + 256)) = p1;
}

// ---------------------------------------------------------------------------
extern "C" void kernel_launch(void* const* d_in, const int* in_sizes, int n_in,
                              void* d_out, int out_size, void* d_ws, size_t ws_size,
                              hipStream_t stream)
{
  const float* q    = (const float*)d_in[0];
  const float* k    = (const float*)d_in[1];
  const float* v    = (const float*)d_in[2];
  const float* Wq   = (const float*)d_in[3];
  const float* bq   = (const float*)d_in[4];
  const float* Wk   = (const float*)d_in[5];
  const float* bk   = (const float*)d_in[6];
  const float* Wv   = (const float*)d_in[7];
  const float* bv   = (const float*)d_in[8];
  const float* Wo   = (const float*)d_in[9];
  const float* bo   = (const float*)d_in[10];
  const float* qn_g = (const float*)d_in[11];
  const float* qn_b = (const float*)d_in[12];
  const float* kn_g = (const float*)d_in[13];
  const float* kn_b = (const float*)d_in[14];

  const long long BT = (long long)BB * TT;   // 16384 rows

  // Workspace partition (all bf16 = ushort elements; all offsets 16B-aligned)
  unsigned short* ws = (unsigned short*)d_ws;
  long long off = 0;
  unsigned short* qn  = ws + off; off += BT * CC;
  unsigned short* kn  = ws + off; off += BT * CC;
  unsigned short* vb  = ws + off; off += BT * CC;
  unsigned short* Wqb = ws + off; off += (long long)HH * CC;
  unsigned short* Wkb = ws + off; off += (long long)HH * CC;
  unsigned short* Wvb = ws + off; off += (long long)HH * CC;
  unsigned short* Wob = ws + off; off += (long long)OO * HH;
  unsigned short* Qb  = ws + off; off += BT * HH;
  unsigned short* Kb  = ws + off; off += BT * HH;
  unsigned short* Vt  = ws + off; off += BT * HH;     // transposed: (HH, BT)
  unsigned short* Pb  = ws + off; off += BT * (long long)TT;
  unsigned short* Rb  = ws + off; off += BT * HH;

  float* out = (float*)d_out;                 // (B,T,O)
  float* wgt = out + BT * OO;                 // (B,T,T)

  const float iscale = 0.044194173824159216f; // 1/sqrt(512)

  // 1) cast weights to bf16
  const int WN = HH * CC;                     // 262144, 4 elems/thread
  cast_bf16<<<dim3(WN / 1024), dim3(256), 0, stream>>>(Wq, Wqb, WN);
  cast_bf16<<<dim3(WN / 1024), dim3(256), 0, stream>>>(Wk, Wkb, WN);
  cast_bf16<<<dim3(WN / 1024), dim3(256), 0, stream>>>(Wv, Wvb, WN);
  cast_bf16<<<dim3(WN / 1024), dim3(256), 0, stream>>>(Wo, Wob, WN);

  // 2) layernorm q,k ; cast v
  ln_cast<<<dim3((unsigned)(BT / 8)), dim3(256), 0, stream>>>(q, qn_g, qn_b, qn, 1);
  ln_cast<<<dim3((unsigned)(BT / 8)), dim3(256), 0, stream>>>(k, kn_g, kn_b, kn, 1);
  ln_cast<<<dim3((unsigned)(BT / 8)), dim3(256), 0, stream>>>(v, nullptr, nullptr, vb, 0);

  // 3) projections (M=BT, N=HH, K=CC)
  dim3 gproj(HH / 128, (unsigned)(BT / 64), 1);
  gemm_bf16<<<gproj, 256, 0, stream>>>(qn, CC, 0, Wqb, CC, 0, bq, 1.0f,
                                       nullptr, Qb, 0, HH, 0, CC);
  gemm_bf16<<<gproj, 256, 0, stream>>>(kn, CC, 0, Wkb, CC, 0, bk, 1.0f,
                                       nullptr, Kb, 0, HH, 0, CC);
  gemm_bf16<<<gproj, 256, 0, stream>>>(vb, CC, 0, Wvb, CC, 0, bv, 1.0f,
                                       nullptr, Vt, 1, (int)BT, 0, CC);  // transposed

  // 4) scores = Q @ K^T * iscale  -> straight into d_out weights region
  dim3 gscore(TT / 128, TT / 64, BB);
  gemm_bf16<<<gscore, 256, 0, stream>>>(Qb, HH, (long long)TT * HH,
                                        Kb, HH, (long long)TT * HH,
                                        nullptr, iscale,
                                        wgt, nullptr, 0, TT, (long long)TT * TT, HH);

  // 5) softmax in place (+ bf16 copy)
  softmax_row<<<dim3((unsigned)BT), dim3(256), 0, stream>>>(wgt, Pb);

  // 6) R = P @ V   (M=TT, N=HH, K=TT per batch; V read via its transpose)
  dim3 gav(HH / 128, TT / 64, BB);
  gemm_bf16<<<gav, 256, 0, stream>>>(Pb, TT, (long long)TT * TT,
                                     Vt, (int)BT, (long long)TT,
                                     nullptr, 1.0f,
                                     nullptr, Rb, 0, HH, (long long)TT * HH, TT);

  // 7) out = R @ Wo^T + bo  (M=BT, N=OO, K=HH)
  dim3 gout(OO / 128, (unsigned)(BT / 64), 1);
  gemm_bf16<<<gout, 256, 0, stream>>>(Rb, HH, 0, Wob, HH, 0, bo, 1.0f,
                                      out, nullptr, 0, OO, 0, HH);
}
